// HebbianLateralConnections_69458211111443
// MI455X (gfx1250) — compile-verified, matrix operand order checked by reading, not run on hardware
//
#include <hip/hip_runtime.h>
#include <hip/hip_bf16.h>

typedef __attribute__((ext_vector_type(16))) __bf16 v16bf;
typedef __attribute__((ext_vector_type(8)))  __bf16 v8bf;
typedef __attribute__((ext_vector_type(8)))  float  v8f;

static constexpr int H   = 2048;
static constexpr int BM  = 128;
static constexpr int BN  = 128;
static constexpr int BK  = 32;
static constexpr int NK  = H / BK;   // 64 K-steps
static constexpr int LDA = BK + 8;   // 40 bf16 = 80B row stride -> conflict-free b128 reads

// ---------------- W build: zero + COO scatter-add ----------------
__global__ void hebb_zero_w(float* __restrict__ W, int n4) {
  int i = blockIdx.x * blockDim.x + threadIdx.x;
  if (i < n4) ((float4*)W)[i] = make_float4(0.f, 0.f, 0.f, 0.f);
}

__global__ void hebb_scatter_w(float* __restrict__ W,
                               const int* __restrict__ idx,
                               const float* __restrict__ vals, int nnz) {
  int c = blockIdx.x * blockDim.x + threadIdx.x;
  if (c < nnz) {
    int r   = idx[c];          // connection_indices[0][c]
    int col = idx[nnz + c];    // connection_indices[1][c]
    atomicAdd(&W[(size_t)r * H + col], vals[c]);
  }
}

// ---------------- helpers ----------------
__device__ __forceinline__ v8bf cvt8(const float4 a, const float4 b) {
  v8bf r;
  r[0] = (__bf16)a.x; r[1] = (__bf16)a.y; r[2] = (__bf16)a.z; r[3] = (__bf16)a.w;
  r[4] = (__bf16)b.x; r[5] = (__bf16)b.y; r[6] = (__bf16)b.z; r[7] = (__bf16)b.w;
  return r;
}

__device__ __forceinline__ v16bf frag_ld(const __bf16* p) {
  // per-lane 16 bf16: K = kh..kh+7 then 16+kh..16+kh+7 (ISA 16-bit A/B layout)
  union { v16bf v; v8bf h[2]; } u;
  u.h[0] = *(const v8bf*)(p);
  u.h[1] = *(const v8bf*)(p + 16);
  return u.v;
}

// ---------------- fused GEMM: out = x + alpha * (x @ W^T) ----------------
__launch_bounds__(256)
__global__ void hebb_wmma_gemm(const float* __restrict__ x,
                               const float* __restrict__ W,
                               const float* __restrict__ alpha_p,
                               float* __restrict__ out) {
  // double-buffered tiles: 4 * 128*40*2B = 40 KB of the 320 KB WGP LDS
  __shared__ __align__(16) __bf16 As[2][BM * LDA];
  __shared__ __align__(16) __bf16 Bs[2][BN * LDA];

  const int tid   = threadIdx.x;
  const int lane  = tid & 31;
  const int wave  = tid >> 5;            // 8 waves
  const int waveM = (wave >> 1) * 32;    // 4 waves down M -> 32 rows each
  const int waveN = (wave & 1)  * 64;    // 2 waves across N -> 64 cols each

  const int blockM = blockIdx.y * BM;
  const int blockN = blockIdx.x * BN;

  // staging: thread t handles row t/2, half (t&1)*16 of the 128x32 tile
  const int srow = tid >> 1;
  const int scol = (tid & 1) * 16;
  const int soff = srow * LDA + scol;

  const float* xg = x + (size_t)(blockM + srow) * H + scol;
  const float* wg = W + (size_t)(blockN + srow) * H + scol;

  v8f acc[2][4];
  #pragma unroll
  for (int s = 0; s < 2; ++s)
    #pragma unroll
    for (int t = 0; t < 4; ++t)
      #pragma unroll
      for (int r = 0; r < 8; ++r)
        acc[s][t][r] = 0.0f;

  const int fr = lane & 15;          // row within 16x16 fragment
  const int kh = (lane >> 4) * 8;    // K sub-half select per ISA layout

  // ---- prologue: stage tile 0 into buffer 0 ----
  {
    float4 a0 = *(const float4*)(xg);
    float4 a1 = *(const float4*)(xg + 4);
    float4 a2 = *(const float4*)(xg + 8);
    float4 a3 = *(const float4*)(xg + 12);
    float4 b0 = *(const float4*)(wg);
    float4 b1 = *(const float4*)(wg + 4);
    float4 b2 = *(const float4*)(wg + 8);
    float4 b3 = *(const float4*)(wg + 12);
    *(v8bf*)(&As[0][soff])     = cvt8(a0, a1);
    *(v8bf*)(&As[0][soff + 8]) = cvt8(a2, a3);
    *(v8bf*)(&Bs[0][soff])     = cvt8(b0, b1);
    *(v8bf*)(&Bs[0][soff + 8]) = cvt8(b2, b3);
  }
  __syncthreads();

  // ---- main loop: one barrier per K-step, ping-pong buffers ----
  for (int i = 0; i < NK; ++i) {
    const int cur = i & 1;
    const int nxt = cur ^ 1;
    const bool has_next = (i + 1) < NK;
    const int kn = has_next ? (i + 1) * BK : 0;

    // issue next tile's global loads early; wait lands at the cvt below
    float4 a0, a1, a2, a3, b0, b1, b2, b3;
    if (has_next) {
      a0 = *(const float4*)(xg + kn);
      a1 = *(const float4*)(xg + kn + 4);
      a2 = *(const float4*)(xg + kn + 8);
      a3 = *(const float4*)(xg + kn + 12);
      b0 = *(const float4*)(wg + kn);
      b1 = *(const float4*)(wg + kn + 4);
      b2 = *(const float4*)(wg + kn + 8);
      b3 = *(const float4*)(wg + kn + 12);
    }

    // fragments from current buffer (reads complete before this iter's barrier)
    v16bf af[2], bf[4];
    #pragma unroll
    for (int s = 0; s < 2; ++s)
      af[s] = frag_ld(&As[cur][(waveM + s * 16 + fr) * LDA + kh]);
    #pragma unroll
    for (int t = 0; t < 4; ++t)
      bf[t] = frag_ld(&Bs[cur][(waveN + t * 16 + fr) * LDA + kh]);

    #pragma unroll
    for (int s = 0; s < 2; ++s)
      #pragma unroll
      for (int t = 0; t < 4; ++t)
        acc[s][t] = __builtin_amdgcn_wmma_f32_16x16x32_bf16(
            false, af[s], false, bf[t], (short)0, acc[s][t], false, false);

    // stage next tile into the other buffer
    if (has_next) {
      *(v8bf*)(&As[nxt][soff])     = cvt8(a0, a1);
      *(v8bf*)(&As[nxt][soff + 8]) = cvt8(a2, a3);
      *(v8bf*)(&Bs[nxt][soff])     = cvt8(b0, b1);
      *(v8bf*)(&Bs[nxt][soff + 8]) = cvt8(b2, b3);
    }
    __syncthreads();
  }

  // ---- epilogue: out = x + alpha * acc, using C/D VGPR layout ----
  const float alpha = *alpha_p;
  const int roff = (lane >> 4) * 8;
  const int gn0  = blockN + waveN + (lane & 15);
  #pragma unroll
  for (int s = 0; s < 2; ++s) {
    #pragma unroll
    for (int t = 0; t < 4; ++t) {
      #pragma unroll
      for (int r = 0; r < 8; ++r) {
        const int gm = blockM + waveM + s * 16 + roff + r;
        const size_t off = (size_t)gm * H + (size_t)(gn0 + t * 16);
        out[off] = x[off] + alpha * acc[s][t][r];
      }
    }
  }
}

extern "C" void kernel_launch(void* const* d_in, const int* in_sizes, int n_in,
                              void* d_out, int out_size, void* d_ws, size_t ws_size,
                              hipStream_t stream) {
  const float* x     = (const float*)d_in[0];
  const int*   idx   = (const int*)  d_in[1];
  const float* vals  = (const float*)d_in[2];
  const float* alpha = (const float*)d_in[3];
  float* W   = (float*)d_ws;     // 2048*2048*4 = 16 MB dense W
  float* out = (float*)d_out;

  const int nnz = in_sizes[2];
  const int M   = in_sizes[0] / H;   // 4*2048 = 8192 rows

  const int n4 = (H * H) / 4;
  hebb_zero_w<<<(n4 + 255) / 256, 256, 0, stream>>>(W, n4);
  hebb_scatter_w<<<(nnz + 255) / 256, 256, 0, stream>>>(W, idx, vals, nnz);

  dim3 grid(H / BN, M / BM);   // 16 x 64 workgroups
  hebb_wmma_gemm<<<grid, 256, 0, stream>>>(x, W, alpha, out);
}
